// constrained_network_32873679683772
// MI455X (gfx1250) — compile-verified, matrix operand order checked by reading, not run on hardware
//
#include <hip/hip_runtime.h>
#include <math.h>

// ---------------------------------------------------------------------------
// Types for CDNA5 WMMA
// ---------------------------------------------------------------------------
typedef __attribute__((ext_vector_type(16))) _Float16 v16h;
typedef __attribute__((ext_vector_type(8)))  float    v8f;

union Frag16 {
  v16h v;
  _Float16 h[16];
  uint4 q[2];
};

// Load a 16x16x32 WMMA operand fragment (A or B) from a [row][K] f16 array.
// Element j<8  -> K = koff + hf*8 + j        (lanes 0-15: K 0..7, lanes 16-31: K 8..15)
// Element j>=8 -> K = koff + 16 + hf*8 + j-8 (lanes 0-15: K 16..23, lanes 16-31: K 24..31)
__device__ __forceinline__ v16h load_frag(const _Float16* base, int nrow,
                                          int rowlen, int koff, int hf) {
  Frag16 f;
  const char* p = (const char*)(base + (long)nrow * rowlen + koff + hf * 8);
  f.q[0] = *(const uint4*)p;
  f.q[1] = *(const uint4*)(p + 32);
  return f.v;
}

__device__ __forceinline__ float siluf(float x) { return x / (1.f + expf(-x)); }

#define WMMA_F16(a, b, c) \
  __builtin_amdgcn_wmma_f32_16x16x32_f16(false, (a), false, (b), (short)0, (c), false, false)

// ---------------------------------------------------------------------------
// Problem constants
// ---------------------------------------------------------------------------
static constexpr int N_ = 10000;
static constexpr int E_ = 320000;
static constexpr int L_ = 4;
static constexpr int MAXT_ = 100;

// ---------------------------------------------------------------------------
// Prep: per-type effective FCTP matrices  Weff[l][t][o][u] = sum_e emb[t][e]*W[l][o][u][e]
// ---------------------------------------------------------------------------
__global__ void weff_kernel(const float* __restrict__ W, const float* __restrict__ emb,
                            float* __restrict__ out, int O, int U, int total) {
  int i = blockIdx.x * blockDim.x + threadIdx.x;
  if (i >= total) return;
  int u = i % U; int r = i / U;
  int o = r % O; r /= O;
  int t = r % MAXT_; int l = r / MAXT_;
  const float* e = emb + t * 32;
  const float* w = W + (((long)l * O + o) * U + u) * 32;
  float acc = 0.f;
#pragma unroll
  for (int k = 0; k < 32; k++) acc += e[k] * w[k];
  out[i] = acc;
}

// Prep: transpose+pad MLP weights to f16 [l][n][k] (k contiguous) for B fragments
__global__ void pack_kernel(const float* __restrict__ W, _Float16* __restrict__ out,
                            int Kin, int Kpad, int Nout, int total) {
  int i = blockIdx.x * blockDim.x + threadIdx.x;
  if (i >= total) return;
  int k = i % Kpad; int r = i / Kpad;
  int n = r % Nout; int l = r / Nout;
  float v = (k < Kin) ? W[((long)l * Kin + k) * Nout + n] : 0.f;
  out[i] = (_Float16)v;
}

// Prep: Newton-Schulz semi-unitary uplift K (16x2)
__global__ void ns_kernel(const float* __restrict__ PU, float* __restrict__ K) {
  if (threadIdx.x != 0 || blockIdx.x != 0) return;
  float k[32];
  float nrm = 0.f;
  for (int i = 0; i < 32; i++) { k[i] = PU[i]; nrm += k[i] * k[i]; }
  nrm = sqrtf(nrm);
  for (int i = 0; i < 32; i++) k[i] /= nrm;
  for (int it = 0; it < 10; it++) {
    float g00 = 0.f, g01 = 0.f, g11 = 0.f;
    for (int u = 0; u < 16; u++) {
      g00 += k[2 * u] * k[2 * u];
      g01 += k[2 * u] * k[2 * u + 1];
      g11 += k[2 * u + 1] * k[2 * u + 1];
    }
    for (int u = 0; u < 16; u++) {
      float a = k[2 * u], b = k[2 * u + 1];
      k[2 * u]     = 1.5f * a - 0.5f * (a * g00 + b * g01);
      k[2 * u + 1] = 1.5f * b - 0.5f * (a * g01 + b * g11);
    }
  }
  for (int i = 0; i < 32; i++) K[i] = k[i];
}

// Init: xbuf=x, ys=ys_old=0, yv=yv_old=K-uplift of x
__global__ void init_kernel(const float* __restrict__ x, const float* __restrict__ K,
                            float* __restrict__ xb, float* __restrict__ ysA,
                            float* __restrict__ ysB, float* __restrict__ yvA,
                            float* __restrict__ yvB) {
  int n = blockIdx.x * blockDim.x + threadIdx.x;
  if (n >= N_) return;
#pragma unroll
  for (int j = 0; j < 6; j++) xb[n * 6 + j] = x[n * 6 + j];
  for (int j = 0; j < 64; j++) { ysA[n * 64 + j] = 0.f; ysB[n * 64 + j] = 0.f; }
  for (int u = 0; u < 16; u++)
#pragma unroll
    for (int m = 0; m < 3; m++) {
      float v = K[u * 2] * x[n * 6 + m] + K[u * 2 + 1] * x[n * 6 + 3 + m];
      yvA[n * 48 + u * 3 + m] = v;
      yvB[n * 48 + u * 3 + m] = v;
    }
}

__global__ void zero_kernel(float* __restrict__ p, int n) {
  int i = blockIdx.x * blockDim.x + threadIdx.x;
  if (i < n) p[i] = 0.f;
}

__global__ void copy_kernel(const float* __restrict__ a, float* __restrict__ b, int n) {
  int i = blockIdx.x * blockDim.x + threadIdx.x;
  if (i < n) b[i] = a[i];
}

// ---------------------------------------------------------------------------
// Per-layer node pre-pass: fs = lin1_s_eff[type] @ ys, fvv = lin1_v_eff[type] @ yv
// ---------------------------------------------------------------------------
__global__ void node_pre_kernel(const int* __restrict__ types,
                                const float* __restrict__ ys, const float* __restrict__ yv,
                                const float* __restrict__ w_s, const float* __restrict__ w_v,
                                float* __restrict__ fs, float* __restrict__ fvv, int total) {
  int i = blockIdx.x * blockDim.x + threadIdx.x;
  if (i >= total) return;
  int n = i / 80, j = i % 80;
  int ty = types[n];
  if (j < 64) {
    const float* wr = w_s + ((long)ty * 64 + j) * 64;
    const float* yr = ys + (long)n * 64;
    float acc = 0.f;
#pragma unroll 8
    for (int u = 0; u < 64; u++) acc += wr[u] * yr[u];
    fs[(long)n * 64 + j] = acc * 0.022097087f;          // 1/sqrt(64*32)
  } else {
    int o = j - 64;
    const float* wr = w_v + ((long)ty * 16 + o) * 16;
    const float* yr = yv + (long)n * 48;
    float a0 = 0.f, a1 = 0.f, a2 = 0.f;
#pragma unroll
    for (int u = 0; u < 16; u++) {
      float w = wr[u];
      a0 += w * yr[u * 3 + 0];
      a1 += w * yr[u * 3 + 1];
      a2 += w * yr[u * 3 + 2];
    }
    fvv[(long)n * 48 + o * 3 + 0] = a0 * 0.044194174f;  // 1/sqrt(16*32)
    fvv[(long)n * 48 + o * 3 + 1] = a1 * 0.044194174f;
    fvv[(long)n * 48 + o * 3 + 2] = a2 * 0.044194174f;
  }
}

// ---------------------------------------------------------------------------
// Fused edge kernel: geometry -> bessel -> WMMA MLP (16->64->64->192)
//                    -> tensor products -> atomic scatter
// 128 threads = 4 waves, 16 edges per wave.
// ---------------------------------------------------------------------------
__global__ void __launch_bounds__(128)
edge_kernel(const float* __restrict__ xbuf,
            const int* __restrict__ esrc, const int* __restrict__ edst,
            const _Float16* __restrict__ W1t, const _Float16* __restrict__ W2t,
            const _Float16* __restrict__ W3t,
            const float* __restrict__ b1, const float* __restrict__ b2,
            const float* __restrict__ b3,
            const float* __restrict__ fs, const float* __restrict__ fvv,
            float* __restrict__ mid_s, float* __restrict__ mid_v) {
  __shared__ __align__(16) _Float16 h_buf[4][16][64];  // hidden activations (f16)
  __shared__ __align__(16) float    w_buf[4][16][192]; // per-edge TP weights
  __shared__ float a_buf[4][16][6];                    // edge attr: ar(3), av(3)
  __shared__ int   sd_buf[4][16][2];                   // src, dst

  const int lane = threadIdx.x & 31;
  const int wv   = threadIdx.x >> 5;
  const int row  = lane & 15;  // local edge (A-matrix row M)
  const int hf   = lane >> 4;  // K-half selector

  const long e = (long)blockIdx.x * 64 + wv * 16 + row;
  const int src = esrc[e];
  const int dst = edst[e];
  sd_buf[wv][row][0] = src;  // both halves write identical values
  sd_buf[wv][row][1] = dst;

  // ---- Phase A: geometry for this lane's half (hf=0: r-part, hf=1: v-part) ----
  float d0 = xbuf[(long)src * 6 + hf * 3 + 0] - xbuf[(long)dst * 6 + hf * 3 + 0];
  float d1 = xbuf[(long)src * 6 + hf * 3 + 1] - xbuf[(long)dst * 6 + hf * 3 + 1];
  float d2 = xbuf[(long)src * 6 + hf * 3 + 2] - xbuf[(long)dst * 6 + hf * 3 + 2];
  float r = sqrtf(d0 * d0 + d1 * d1 + d2 * d2);
  float tt = r * 0.4f;                       // r / MR
  float uu = 2.f * (tt - 1.f);
  float cy = (1.f - cosf(3.14159265358979f * uu)) * 0.5f;
  float cut = (uu > 0.f) ? 0.f : ((uu < -1.f) ? 1.f : cy);
  float sca = cut * 1.7320508075688772f / r; // cutoff * sqrt(3)/r
  a_buf[wv][row][hf * 3 + 0] = sca * d0;
  a_buf[wv][row][hf * 3 + 1] = sca * d1;
  a_buf[wv][row][hf * 3 + 2] = sca * d2;

  // Bessel features, built directly in A-fragment register layout
  float kb  = 3.14159265358979f * r * 0.4f;
  float amp = 2.5298221281347035f / r;       // sqrt(2/MR)*sqrt(NB)
  Frag16 af;
#pragma unroll
  for (int k = 0; k < 8; k++) af.h[k] = (_Float16)(amp * sinf((float)(k + 1) * kb));
#pragma unroll
  for (int k = 8; k < 16; k++) af.h[k] = (_Float16)0.f;

  // ---- GEMM1: ef(16x32 padded) @ W1(32x64) + b1, silu ----
#pragma unroll
  for (int t = 0; t < 4; t++) {
    int n = t * 16 + row;
    v16h bf = load_frag(W1t, n, 32, 0, hf);
    float bb = b1[n];
    v8f c = {bb, bb, bb, bb, bb, bb, bb, bb};
    c = WMMA_F16(af.v, bf, c);
#pragma unroll
    for (int g = 0; g < 8; g++)
      h_buf[wv][g + hf * 8][n] = (_Float16)siluf(c[g]);
  }

  // ---- GEMM2: h1(16x64) @ W2(64x64) + b2, silu ----
  {
    v16h a0 = load_frag(&h_buf[wv][0][0], row, 64, 0, hf);
    v16h a1 = load_frag(&h_buf[wv][0][0], row, 64, 32, hf);
#pragma unroll
    for (int t = 0; t < 4; t++) {
      int n = t * 16 + row;
      v16h q0 = load_frag(W2t, n, 64, 0, hf);
      v16h q1 = load_frag(W2t, n, 64, 32, hf);
      float bb = b2[n];
      v8f c = {bb, bb, bb, bb, bb, bb, bb, bb};
      c = WMMA_F16(a0, q0, c);
      c = WMMA_F16(a1, q1, c);
#pragma unroll
      for (int g = 0; g < 8; g++)
        h_buf[wv][g + hf * 8][n] = (_Float16)siluf(c[g]);
    }
  }

  // ---- GEMM3: h2(16x64) @ W3(64x192) + b3 -> w_buf ----
  {
    v16h a0 = load_frag(&h_buf[wv][0][0], row, 64, 0, hf);
    v16h a1 = load_frag(&h_buf[wv][0][0], row, 64, 32, hf);
#pragma unroll
    for (int t = 0; t < 12; t++) {
      int n = t * 16 + row;
      v16h q0 = load_frag(W3t, n, 64, 0, hf);
      v16h q1 = load_frag(W3t, n, 64, 32, hf);
      float bb = b3[n];
      v8f c = {bb, bb, bb, bb, bb, bb, bb, bb};
      c = WMMA_F16(a0, q0, c);
      c = WMMA_F16(a1, q1, c);
#pragma unroll
      for (int g = 0; g < 8; g++)
        w_buf[wv][g + hf * 8][n] = c[g];
    }
  }

  // ---- Tensor products + atomic scatter (norms folded: /sqrt(DEG)=1/sqrt(32)) ----
  // o1: 0e x 1o -> 1o.  scale = 1/sqrt(2)/sqrt(32) = 0.125
  for (int it = lane; it < 1024; it += 32) {
    int el = it >> 6, u = it & 63;
    int s_ = sd_buf[wv][el][0], d_ = sd_buf[wv][el][1];
    float sv = fs[(long)s_ * 64 + u];
    float w0 = w_buf[wv][el][2 * u], w1v = w_buf[wv][el][2 * u + 1];
    float c0 = 0.125f * sv;
#pragma unroll
    for (int m = 0; m < 3; m++) {
      float val = c0 * (w0 * a_buf[wv][el][m] + w1v * a_buf[wv][el][3 + m]);
      atomicAdd(&mid_v[((long)d_ * 80 + u) * 3 + m], val);
    }
  }
  // o2: 1o.1o -> 0e (scale 1/sqrt(6)/sqrt(32)); o3: cross -> 1o (scale 0.5/sqrt(32))
  for (int it = lane; it < 256; it += 32) {
    int el = it >> 4, u = it & 15;
    int s_ = sd_buf[wv][el][0], d_ = sd_buf[wv][el][1];
    float v0 = fvv[(long)s_ * 48 + u * 3 + 0];
    float v1 = fvv[(long)s_ * 48 + u * 3 + 1];
    float v2 = fvv[(long)s_ * 48 + u * 3 + 2];
    float ar0 = a_buf[wv][el][0], ar1 = a_buf[wv][el][1], ar2 = a_buf[wv][el][2];
    float av0 = a_buf[wv][el][3], av1 = a_buf[wv][el][4], av2 = a_buf[wv][el][5];
    float dot0 = v0 * ar0 + v1 * ar1 + v2 * ar2;
    float dot1 = v0 * av0 + v1 * av1 + v2 * av2;
    float w20 = w_buf[wv][el][128 + 2 * u], w21 = w_buf[wv][el][129 + 2 * u];
    atomicAdd(&mid_s[(long)d_ * 16 + u], 0.07216878f * (w20 * dot0 + w21 * dot1));
    float cr00 = v1 * ar2 - v2 * ar1, cr01 = v2 * ar0 - v0 * ar2, cr02 = v0 * ar1 - v1 * ar0;
    float cr10 = v1 * av2 - v2 * av1, cr11 = v2 * av0 - v0 * av2, cr12 = v0 * av1 - v1 * av0;
    float w30 = w_buf[wv][el][160 + 2 * u], w31 = w_buf[wv][el][161 + 2 * u];
    const float s3 = 0.08838835f;
    atomicAdd(&mid_v[((long)d_ * 80 + 64 + u) * 3 + 0], s3 * (w30 * cr00 + w31 * cr10));
    atomicAdd(&mid_v[((long)d_ * 80 + 64 + u) * 3 + 1], s3 * (w30 * cr01 + w31 * cr11));
    atomicAdd(&mid_v[((long)d_ * 80 + 64 + u) * 3 + 2], s3 * (w30 * cr02 + w31 * cr12));
  }
}

// ---------------------------------------------------------------------------
// Per-layer node post-pass: lin2 + self-connection + gate + self-interaction
// + leapfrog update + projection to x.  One block per node.
// Writes new state into ys_old / yv_old buffers (host swaps pointers).
// ---------------------------------------------------------------------------
__global__ void __launch_bounds__(128)
node_post_kernel(const int* __restrict__ types,
                 const float* __restrict__ ys, const float* __restrict__ yv,
                 float* __restrict__ ys_old, float* __restrict__ yv_old,
                 const float* __restrict__ mid_s, const float* __restrict__ mid_v,
                 const float* __restrict__ w_sc_s, const float* __restrict__ w_sc_v,
                 const float* __restrict__ w_l2_s, const float* __restrict__ w_l2_v,
                 const float* __restrict__ si_Ws, const float* __restrict__ si_Wv,
                 const float* __restrict__ hstep, const float* __restrict__ mixp,
                 const float* __restrict__ Kmat, float* __restrict__ xout) {
  const int n = blockIdx.x;
  const int tid = threadIdx.x;
  const int ty = types[n];
  __shared__ float ys_l[64], yv_l[48], convs[80], convv[48], nv_l[48];
  if (tid < 64) ys_l[tid] = ys[(long)n * 64 + tid];
  if (tid < 48) yv_l[tid] = yv[(long)n * 48 + tid];
  __syncthreads();
  const float h2 = hstep[0] * hstep[0];
  const float mx = mixp[0];
  const float c_s = 0.38268343f, c_x = 0.92387953f;  // sin(pi/8), cos(pi/8)
  if (tid < 80) {
    const float* wr = w_sc_s + ((long)ty * 80 + tid) * 64;
    float acc = 0.f;
#pragma unroll 8
    for (int u = 0; u < 64; u++) acc += wr[u] * ys_l[u];
    float scs = acc * 0.022097087f;                       // 1/sqrt(64*32)
    const float* w2r = w_l2_s + ((long)ty * 80 + tid) * 16;
    float acc2 = 0.f;
#pragma unroll
    for (int u = 0; u < 16; u++) acc2 += w2r[u] * mid_s[(long)n * 16 + u];
    float cs2 = acc2 * 0.044194174f;                      // 1/sqrt(16*32)
    convs[tid] = c_s * scs + c_x * cs2;
  }
  if (tid < 48) {
    int o = tid / 3, m = tid % 3;
    const float* wr = w_sc_v + ((long)ty * 16 + o) * 16;
    float acc = 0.f;
#pragma unroll
    for (int u = 0; u < 16; u++) acc += wr[u] * yv_l[u * 3 + m];
    float scv = acc * 0.044194174f;                       // 1/sqrt(16*32)
    const float* w2r = w_l2_v + ((long)ty * 16 + o) * 80;
    float acc2 = 0.f;
#pragma unroll 8
    for (int i = 0; i < 80; i++) acc2 += w2r[i] * mid_v[((long)n * 80 + i) * 3 + m];
    float cv2 = acc2 * 0.019764235f;                      // 1/sqrt(80*32)
    convv[tid] = c_s * scv + c_x * cv2;
  }
  __syncthreads();
  if (tid < 64) {
    float g = convs[tid];
    float gns = g / (1.f + expf(-g));
    const float* sw = si_Ws + tid * 64;
    float acc = 0.f;
#pragma unroll 8
    for (int u = 0; u < 64; u++) acc += sw[u] * ys_l[u];
    float sis = acc * 0.125f;                             // 1/sqrt(64)
    float ns = 2.f * ys_l[tid] - ys_old[(long)n * 64 + tid] +
               h2 * (mx * gns + (mx - 1.f) * sis);
    ys_old[(long)n * 64 + tid] = ns;
  }
  if (tid < 48) {
    int o = tid / 3, m = tid % 3;
    float gate = 1.f / (1.f + expf(-convs[64 + o]));
    float gnv = gate * convv[tid];
    const float* sw = si_Wv + o * 16;
    float acc = 0.f;
#pragma unroll
    for (int u = 0; u < 16; u++) acc += sw[u] * yv_l[u * 3 + m];
    float siv = acc * 0.25f;                              // 1/sqrt(16)
    float nv = 2.f * yv_l[tid] - yv_old[(long)n * 48 + tid] +
               h2 * (mx * gnv + (mx - 1.f) * siv);
    yv_old[(long)n * 48 + tid] = nv;
    nv_l[tid] = nv;
  }
  __syncthreads();
  if (tid < 6) {
    int i = tid / 3, m = tid % 3;
    float acc = 0.f;
#pragma unroll
    for (int u = 0; u < 16; u++) acc += Kmat[u * 2 + i] * nv_l[u * 3 + m];
    xout[(long)n * 6 + tid] = acc;
  }
}

// ---------------------------------------------------------------------------
// Host side
// ---------------------------------------------------------------------------
extern "C" void kernel_launch(void* const* d_in, const int* in_sizes, int n_in,
                              void* d_out, int out_size, void* d_ws, size_t ws_size,
                              hipStream_t stream) {
  (void)in_sizes; (void)n_in; (void)out_size; (void)ws_size;
  const float* x_in   = (const float*)d_in[0];
  const int*   types  = (const int*)d_in[2];
  const int*   esrc   = (const int*)d_in[3];
  const int*   edst   = (const int*)d_in[4];
  const float* emb    = (const float*)d_in[5];
  const float* PU     = (const float*)d_in[6];
  const float* hstep  = (const float*)d_in[7];
  const float* mixp   = (const float*)d_in[8];
  const float* si_Ws  = (const float*)d_in[9];
  const float* si_Wv  = (const float*)d_in[10];
  const float* sc_Ws  = (const float*)d_in[11];
  const float* sc_Wv  = (const float*)d_in[12];
  const float* l1_Ws  = (const float*)d_in[13];
  const float* l1_Wv  = (const float*)d_in[14];
  const float* l2_Ws  = (const float*)d_in[15];
  const float* l2_Wv  = (const float*)d_in[16];
  const float* fcW1   = (const float*)d_in[17];
  const float* fcb1   = (const float*)d_in[18];
  const float* fcW2   = (const float*)d_in[19];
  const float* fcb2   = (const float*)d_in[20];
  const float* fcW3   = (const float*)d_in[21];
  const float* fcb3   = (const float*)d_in[22];

  char* ws = (char*)d_ws;
  size_t cur = 0;
  auto alloc = [&](size_t bytes) -> char* {
    char* p = ws + cur;
    cur += (bytes + 255) & ~(size_t)255;
    return p;
  };

  float* Kmat = (float*)alloc(32 * 4);
  float* xb   = (float*)alloc((size_t)N_ * 6 * 4);
  float* ysA  = (float*)alloc((size_t)N_ * 64 * 4);
  float* ysB  = (float*)alloc((size_t)N_ * 64 * 4);
  float* yvA  = (float*)alloc((size_t)N_ * 48 * 4);
  float* yvB  = (float*)alloc((size_t)N_ * 48 * 4);
  float* fs   = (float*)alloc((size_t)N_ * 64 * 4);
  float* fvv  = (float*)alloc((size_t)N_ * 48 * 4);
  float* mids = (float*)alloc((size_t)N_ * 16 * 4);
  float* midv = (float*)alloc((size_t)N_ * 240 * 4);
  float* l1s_eff = (float*)alloc((size_t)L_ * MAXT_ * 64 * 64 * 4);
  float* l1v_eff = (float*)alloc((size_t)L_ * MAXT_ * 16 * 16 * 4);
  float* l2s_eff = (float*)alloc((size_t)L_ * MAXT_ * 80 * 16 * 4);
  float* l2v_eff = (float*)alloc((size_t)L_ * MAXT_ * 16 * 80 * 4);
  float* scs_eff = (float*)alloc((size_t)L_ * MAXT_ * 80 * 64 * 4);
  float* scv_eff = (float*)alloc((size_t)L_ * MAXT_ * 16 * 16 * 4);
  _Float16* W1t = (_Float16*)alloc((size_t)L_ * 64 * 32 * 2);
  _Float16* W2t = (_Float16*)alloc((size_t)L_ * 64 * 64 * 2);
  _Float16* W3t = (_Float16*)alloc((size_t)L_ * 192 * 64 * 2);

  auto grid1 = [](int total) { return (total + 255) / 256; };

  // ---- Prep ----
  weff_kernel<<<grid1(L_ * MAXT_ * 64 * 64), 256, 0, stream>>>(l1_Ws, emb, l1s_eff, 64, 64, L_ * MAXT_ * 64 * 64);
  weff_kernel<<<grid1(L_ * MAXT_ * 16 * 16), 256, 0, stream>>>(l1_Wv, emb, l1v_eff, 16, 16, L_ * MAXT_ * 16 * 16);
  weff_kernel<<<grid1(L_ * MAXT_ * 80 * 16), 256, 0, stream>>>(l2_Ws, emb, l2s_eff, 80, 16, L_ * MAXT_ * 80 * 16);
  weff_kernel<<<grid1(L_ * MAXT_ * 16 * 80), 256, 0, stream>>>(l2_Wv, emb, l2v_eff, 16, 80, L_ * MAXT_ * 16 * 80);
  weff_kernel<<<grid1(L_ * MAXT_ * 80 * 64), 256, 0, stream>>>(sc_Ws, emb, scs_eff, 80, 64, L_ * MAXT_ * 80 * 64);
  weff_kernel<<<grid1(L_ * MAXT_ * 16 * 16), 256, 0, stream>>>(sc_Wv, emb, scv_eff, 16, 16, L_ * MAXT_ * 16 * 16);
  pack_kernel<<<grid1(L_ * 64 * 32), 256, 0, stream>>>(fcW1, W1t, 16, 32, 64, L_ * 64 * 32);
  pack_kernel<<<grid1(L_ * 64 * 64), 256, 0, stream>>>(fcW2, W2t, 64, 64, 64, L_ * 64 * 64);
  pack_kernel<<<grid1(L_ * 192 * 64), 256, 0, stream>>>(fcW3, W3t, 64, 64, 192, L_ * 192 * 64);
  ns_kernel<<<1, 32, 0, stream>>>(PU, Kmat);
  init_kernel<<<(N_ + 127) / 128, 128, 0, stream>>>(x_in, Kmat, xb, ysA, ysB, yvA, yvB);

  float* ys = ysA; float* ys_old = ysB;
  float* yv = yvA; float* yv_old = yvB;

  // ---- Layers ----
  for (int l = 0; l < L_; l++) {
    zero_kernel<<<grid1(N_ * 16), 256, 0, stream>>>(mids, N_ * 16);
    zero_kernel<<<grid1(N_ * 240), 256, 0, stream>>>(midv, N_ * 240);
    node_pre_kernel<<<grid1(N_ * 80), 256, 0, stream>>>(
        types, ys, yv,
        l1s_eff + (size_t)l * MAXT_ * 64 * 64,
        l1v_eff + (size_t)l * MAXT_ * 16 * 16,
        fs, fvv, N_ * 80);
    edge_kernel<<<E_ / 64, 128, 0, stream>>>(
        xb, esrc, edst,
        W1t + (size_t)l * 64 * 32,
        W2t + (size_t)l * 64 * 64,
        W3t + (size_t)l * 192 * 64,
        fcb1 + (size_t)l * 64, fcb2 + (size_t)l * 64, fcb3 + (size_t)l * 192,
        fs, fvv, mids, midv);
    node_post_kernel<<<N_, 128, 0, stream>>>(
        types, ys, yv, ys_old, yv_old, mids, midv,
        scs_eff + (size_t)l * MAXT_ * 80 * 64,
        scv_eff + (size_t)l * MAXT_ * 16 * 16,
        l2s_eff + (size_t)l * MAXT_ * 80 * 16,
        l2v_eff + (size_t)l * MAXT_ * 16 * 80,
        si_Ws + (size_t)l * 64 * 64,
        si_Wv + (size_t)l * 16 * 16,
        hstep + l, mixp + l, Kmat, xb);
    float* t;
    t = ys; ys = ys_old; ys_old = t;
    t = yv; yv = yv_old; yv_old = t;
  }

  copy_kernel<<<grid1(N_ * 6), 256, 0, stream>>>(xb, (float*)d_out, N_ * 6);
}
